// PerformerProjection_64656437674193
// MI455X (gfx1250) — matrix operand order repaired and checked by reading, not compiled, not run on hardware
//
#include <hip/hip_runtime.h>

typedef __bf16 bf16;
typedef __attribute__((ext_vector_type(16))) __bf16 v16bf;
typedef __attribute__((ext_vector_type(8)))  __bf16 v8bf;
typedef __attribute__((ext_vector_type(8)))  float  v8f;
typedef __attribute__((ext_vector_type(4)))  float  vf4;
typedef unsigned int v4u __attribute__((ext_vector_type(4)));
typedef int          v8i __attribute__((ext_vector_type(8)));
typedef int          v4i __attribute__((ext_vector_type(4)));

namespace {
constexpr int   BH   = 64;      // B*H
constexpr int   L    = 4096;
constexpr int   D    = 64;
constexpr int   M    = 266;     // random features
constexpr int   MT   = 17;      // ceil(M/16)
constexpr int   MP   = 272;     // MT*16
constexpr int   MK   = 288;     // M padded to 9 K-steps of 32
constexpr int   CH   = 128;     // L-chunk rows per pass
constexpr int   PL   = 136;     // kpT / vT row pitch (elements)
constexpr float EPSF = 1e-3f;
constexpr int   SPLITS       = 8;
constexpr int   CH_PER_SPLIT = (L / CH) / SPLITS;   // 4
constexpr int   PART_STRIDE  = MP * D + MP;         // 17680 f32 per partial slab
constexpr int   KVT_ELEMS    = 80 * MK;             // 23040 bf16 per kv_ext slab

// ---- K1 LDS layout (bytes), all 16B aligned ----
constexpr int O1_PK   = 0;                              // packed proj B-frags: 17*2*32*16 bf16
constexpr int O1_XROW = O1_PK + MT * 2 * 32 * 16 * 2;   // 34816  : k rows, 128x64 bf16
constexpr int O1_KSUM = O1_XROW + CH * D * 2;           // 51200  : 272 f32
constexpr int O1_VT   = O1_KSUM + MP * 4;               // 52288  : v^T 64x136 bf16
constexpr int O1_KPT  = O1_VT + D * PL * 2;             // 69696  : kp^T 272x136 bf16
constexpr int LDS1    = O1_KPT + MP * PL * 2;           // 143680

// ---- K3 LDS layout ----
constexpr int O3_PK   = 0;
constexpr int O3_XROW = O3_PK + MT * 2 * 32 * 16 * 2;   // 34816  : q rows
constexpr int O3_KVT  = O3_XROW + CH * D * 2;           // 51200  : kv_ext 80x288 bf16
constexpr int O3_QP   = O3_KVT + KVT_ELEMS * 2;         // 97280  : qp 128x288 bf16
constexpr int LDS3    = O3_QP + CH * MK * 2;            // 171008
} // namespace

__device__ __forceinline__ v8f zero8() { v8f z = {0,0,0,0,0,0,0,0}; return z; }

// A-matrix fragment (16-bit, 16x32): lane holds row M=lane&15; K linear:
// elems 0..15 -> K = kbase + 16*hi + j, with klo = kbase + 8*hi.
__device__ __forceinline__ v16bf ldfragA(const bf16* p, int klo) {
  union { v16bf v; v8bf h[2]; } u;
  u.h[0] = *(const v8bf*)(p + klo + (klo & 31));
  u.h[1] = *(const v8bf*)(p + klo + (klo & 31) + 8);
  return u.v;
}
// B-matrix fragment (16-bit, 32x16): lane holds col N=lane&15; K linear:
// elems 0..15 -> K = kbase + 16*hi + j  (contiguous).
__device__ __forceinline__ v16bf ldfragB(const bf16* p, int kb) {
  union { v16bf v; v8bf h[2]; } u;
  u.h[0] = *(const v8bf*)(p + kb);
  u.h[1] = *(const v8bf*)(p + kb + 8);
  return u.v;
}

__device__ __forceinline__ v8f wmma32(v16bf a, v16bf b, v8f c) {
  return __builtin_amdgcn_wmma_f32_16x16x32_bf16(false, a, false, b, (short)0, c,
                                                 false, false);
}

// Pack proj (global f32 [266][64], row m = feature) into per-lane contiguous
// B-fragments: pk[(mt*2+ks)*32 + lane][16], elem j = proj[m][ks*32 + 16*hi + j].
__device__ __forceinline__ void pack_proj(bf16* pk, const float* __restrict__ proj,
                                          int t) {
  for (int e = t; e < MT * 2 * 32; e += 256) {
    int el  = e & 31;
    int ks2 = (e >> 5) & 1;
    int mt  = e >> 6;
    int m   = mt * 16 + (el & 15);
    int kb  = ks2 * 32 + (el >> 4) * 16;
    bf16* dst = pk + e * 16;
    if (m < M) {
#pragma unroll
      for (int j = 0; j < 16; ++j) dst[j] = (bf16)proj[m * D + kb + j];
    } else {
#pragma unroll
      for (int j = 0; j < 16; ++j) dst[j] = (bf16)0.0f;
    }
  }
}

// ---------------- K1: kp = relu(K proj^T)+eps ; partial k_sum, kv ----------------
__global__ __launch_bounds__(256)
void k1_features_kv(const float* __restrict__ kg, const float* __restrict__ vg,
                    const float* __restrict__ proj, float* __restrict__ part) {
  extern __shared__ char smem[];
  bf16*  pk   = (bf16*)(smem + O1_PK);
  bf16*  xrow = (bf16*)(smem + O1_XROW);
  float* ksum = (float*)(smem + O1_KSUM);
  bf16*  vT   = (bf16*)(smem + O1_VT);
  bf16*  kpT  = (bf16*)(smem + O1_KPT);

  const int t = threadIdx.x, lane = t & 31, w = t >> 5;
  const int n16 = lane & 15, hi = lane >> 4;
  const int bh = blockIdx.x, s = blockIdx.y;

  const float* kb = kg + ((size_t)bh * L + (size_t)s * CH_PER_SPLIT * CH) * D;
  const float* vb = vg + ((size_t)bh * L + (size_t)s * CH_PER_SPLIT * CH) * D;
  float* pslab = part + (size_t)(bh * SPLITS + s) * PART_STRIDE;

  pack_proj(pk, proj, t);
  for (int m = t; m < MP; m += 256) ksum[m] = 0.0f;

  v8f kvacc[9];
#pragma unroll
  for (int i = 0; i < 9; ++i) kvacc[i] = zero8();
  __syncthreads();

  for (int c = 0; c < CH_PER_SPLIT; ++c) {
    const float* kc = kb + (size_t)c * CH * D;
    const float* vc = vb + (size_t)c * CH * D;

    // k chunk -> LDS row-major bf16 (coalesced float4 loads)
    for (int i = 0; i < 8; ++i) {
      int idx = i * 256 + t;
      vf4 x = *(const vf4*)(kc + (size_t)idx * 4);
      bf16* d = xrow + idx * 4;
      d[0] = (bf16)x.x; d[1] = (bf16)x.y; d[2] = (bf16)x.z; d[3] = (bf16)x.w;
    }
    // v chunk -> LDS transposed vT[d][l]
    {
      int dc = t & 63, seg = t >> 6;
      for (int i = 0; i < 32; ++i) {
        int l = seg * 32 + i;
        vT[dc * PL + l] = (bf16)vc[(size_t)l * D + dc];
      }
    }
    if (c + 1 < CH_PER_SPLIT) {
      __builtin_prefetch(kc + CH * D + t * 32, 0, 1);
      __builtin_prefetch(vc + CH * D + t * 32, 0, 1);
    }
    __syncthreads();

    // GEMM1: kp tile (wave's 16 rows x 17 m-tiles), store kp^T packed
    {
      const bf16* arow = xrow + (w * 16 + n16) * D;
      v16bf a0 = ldfragA(arow, hi * 8);
      v16bf a1 = ldfragA(arow, 32 + hi * 8);
      for (int mt = 0; mt < MT; ++mt) {
        v16bf b0 = *(const v16bf*)(pk + ((mt * 2 + 0) * 32 + lane) * 16);
        v16bf b1 = *(const v16bf*)(pk + ((mt * 2 + 1) * 32 + lane) * 16);
        v8f acc = zero8();
        acc = wmma32(a0, b0, acc);
        acc = wmma32(a1, b1, acc);
        int m = mt * 16 + n16;            // lane holds one kp column = kp^T row
        float ep = (m < M) ? EPSF : 0.0f; // padded features stay exactly zero
        v8bf p;
#pragma unroll
        for (int r = 0; r < 8; ++r) p[r] = (bf16)(fmaxf(acc[r], 0.0f) + ep);
        *(v8bf*)(kpT + m * PL + w * 16 + hi * 8) = p;   // one b128 store
      }
    }
    __syncthreads();

    // k_sum accumulation (one owner thread per m)
    for (int m = t; m < MP; m += 256) {
      float ssum = 0.0f;
      const bf16* row = kpT + m * PL;
      for (int l = 0; l < CH; ++l) ssum += (float)row[l];
      ksum[m] += ssum;
    }

    // GEMM3: kv[m][d] += kp^T * v   (68 tiles distributed over 8 waves)
    {
      int i = 0;
      for (int tid = w; tid < MT * 4; tid += 8, ++i) {
        int mt = tid >> 2, dt = tid & 3;
        const bf16* ar = kpT + (mt * 16 + n16) * PL;
        const bf16* br = vT + (dt * 16 + n16) * PL;
        v8f acc = kvacc[i];
#pragma unroll
        for (int ks = 0; ks < 4; ++ks)
          acc = wmma32(ldfragA(ar, ks * 32 + hi * 8),
                       ldfragB(br, ks * 32 + hi * 16), acc);
        kvacc[i] = acc;
      }
    }
    __syncthreads();
  }

  // write f32 partial kv + k_sum
  {
    int i = 0;
    for (int tid = w; tid < MT * 4; tid += 8, ++i) {
      int mt = tid >> 2, dt = tid & 3;
#pragma unroll
      for (int r = 0; r < 8; ++r) {
        int m = mt * 16 + hi * 8 + r;
        pslab[(size_t)m * D + dt * 16 + n16] = kvacc[i][r];
      }
    }
  }
  for (int m = t; m < MP; m += 256) pslab[(size_t)MP * D + m] = ksum[m];
}

// ---------------- K2: reduce partials -> kv_ext bf16 slab [80][288] ----------------
__global__ __launch_bounds__(256)
void k2_reduce(const float* __restrict__ part, bf16* __restrict__ kvt) {
  const int bh = blockIdx.x, t = threadIdx.x;
  const float* pb = part + (size_t)bh * SPLITS * PART_STRIDE;
  bf16* dst = kvt + (size_t)bh * KVT_ELEMS;
  for (int e = t; e < KVT_ELEMS; e += 256) {
    int d = e / MK, m = e - d * MK;
    float ssum = 0.0f;
    if (m < MP) {
      if (d < D) {
#pragma unroll
        for (int s = 0; s < SPLITS; ++s)
          ssum += pb[(size_t)s * PART_STRIDE + (size_t)m * D + d];
      } else if (d == D) {   // row 64 = k_sum -> denominator column of GEMM2
#pragma unroll
        for (int s = 0; s < SPLITS; ++s)
          ssum += pb[(size_t)s * PART_STRIDE + (size_t)MP * D + m];
      }
    }
    dst[e] = (bf16)ssum;     // rows 65..79 and m>=272 are exactly zero
  }
}

// ---------------- K3: qp = relu(Q proj^T)+eps ; out = (qp kv) / (qp k_sum) ----------------
__global__ __launch_bounds__(256)
void k3_output(const float* __restrict__ qg, const float* __restrict__ proj,
               const bf16* __restrict__ kvt, float* __restrict__ outg) {
  extern __shared__ char smem[];
  bf16* pk   = (bf16*)(smem + O3_PK);
  bf16* xrow = (bf16*)(smem + O3_XROW);
  bf16* kvT  = (bf16*)(smem + O3_KVT);
  bf16* qp   = (bf16*)(smem + O3_QP);

  const int t = threadIdx.x, lane = t & 31, w = t >> 5;
  const int n16 = lane & 15, hi = lane >> 4;
  const int bh = blockIdx.x, c = blockIdx.y;

#if defined(__gfx1250__) && __has_builtin(__builtin_amdgcn_tensor_load_to_lds) && \
    __has_builtin(__builtin_amdgcn_s_wait_tensorcnt)
#define USE_TDM 1
#else
#define USE_TDM 0
#endif

#if USE_TDM
  // ---- TDM: DMA the 80x288 bf16 kv_ext slab global -> LDS, wave 0 issues ----
  if (w == 0) {
    unsigned long long ga =
        (unsigned long long)(const void*)(kvt + (size_t)bh * KVT_ELEMS);
    unsigned int lds_off = (unsigned int)(size_t)(void*)kvT;  // addr[31:0] = LDS offset
    v4u g0;
    g0.x = 1u;                                        // count=1 (valid user D#)
    g0.y = lds_off;                                   // lds_addr [63:32]
    g0.z = (unsigned int)(ga & 0xFFFFFFFFu);          // global_addr[31:0]
    g0.w = (unsigned int)((ga >> 32) & 0x1FFFFFFu)    // global_addr[56:32]
           | (2u << 30);                              // type = 2 ("image")
    v8i g1;
    g1[0] = (int)(1u << 16);          // workgroup_mask=0 | data_size=1 (2 bytes)
    g1[1] = (int)(288u << 16);        // tensor_dim0[15:0]=288 in bits[63:48]
    g1[2] = (int)(80u << 16);         // tensor_dim0[31:16]=0 | tensor_dim1[15:0]=80
    g1[3] = (int)(288u << 16);        // tensor_dim1[31:16]=0 | tile_dim0=288
    g1[4] = 80;                       // tile_dim1=80 | tile_dim2=0
    g1[5] = 288;                      // tensor_dim0_stride[31:0]=288
    g1[6] = 0;                        // stride0[47:32]=0 | stride1[15:0]=0
    g1[7] = 0;
    v4i gz4 = {0, 0, 0, 0};           // groups 2/3 unused (<=2D tensor)
    v8i gz8 = {0, 0, 0, 0, 0, 0, 0, 0};
    __builtin_amdgcn_tensor_load_to_lds(g0, g1, gz4, gz4, gz8, 0);
  }
#endif

  pack_proj(pk, proj, t);
#if !USE_TDM
  {
    // fallback: staged copy through VGPRs
    const unsigned int* src = (const unsigned int*)(kvt + (size_t)bh * KVT_ELEMS);
    unsigned int* d = (unsigned int*)kvT;
    for (int e = t; e < KVT_ELEMS / 2; e += 256) d[e] = src[e];
  }
#endif
  // zero qp pad columns (m in [272,288))
  for (int e = t; e < CH * 16; e += 256)
    qp[(e >> 4) * MK + MP + (e & 15)] = (bf16)0.0f;

  const float* qc = qg + ((size_t)bh * L + (size_t)c * CH) * D;
  for (int i = 0; i < 8; ++i) {
    int idx = i * 256 + t;
    vf4 x = *(const vf4*)(qc + (size_t)idx * 4);
    bf16* dd = xrow + idx * 4;
    dd[0] = (bf16)x.x; dd[1] = (bf16)x.y; dd[2] = (bf16)x.z; dd[3] = (bf16)x.w;
  }

#if USE_TDM
  if (w == 0) __builtin_amdgcn_s_wait_tensorcnt(0);   // slab DMA complete
#endif
  __syncthreads();

  // qp tiles -> LDS row-major [l][m]
  {
    const bf16* arow = xrow + (w * 16 + n16) * D;
    v16bf a0 = ldfragA(arow, hi * 8);
    v16bf a1 = ldfragA(arow, 32 + hi * 8);
    for (int mt = 0; mt < MT; ++mt) {
      v16bf b0 = *(const v16bf*)(pk + ((mt * 2 + 0) * 32 + lane) * 16);
      v16bf b1 = *(const v16bf*)(pk + ((mt * 2 + 1) * 32 + lane) * 16);
      v8f acc = zero8();
      acc = wmma32(a0, b0, acc);
      acc = wmma32(a1, b1, acc);
      int m = mt * 16 + n16;
      float ep = (m < M) ? EPSF : 0.0f;
#pragma unroll
      for (int r = 0; r < 8; ++r)
        qp[(w * 16 + hi * 8 + r) * MK + m] = (bf16)(fmaxf(acc[r], 0.0f) + ep);
    }
  }
  __syncthreads();

  // GEMM2: qkv_ext = qp * kv_ext  (5 d-tiles, tile 4 col 64 = denominator)
  {
    v8f o[5];
#pragma unroll
    for (int i = 0; i < 5; ++i) o[i] = zero8();
    const bf16* arow = qp + (w * 16 + n16) * MK;
    for (int ks = 0; ks < 9; ++ks) {
      v16bf a = ldfragA(arow, ks * 32 + hi * 8);
#pragma unroll
      for (int dt = 0; dt < 5; ++dt) {
        v16bf b = ldfragB(kvT + (dt * 16 + n16) * MK, ks * 32 + hi * 16);
        o[dt] = wmma32(a, b, o[dt]);
      }
    }
    // denominator lives in lanes 0 (rows 0..7) / 16 (rows 8..15) of tile 4
    int srcb = (lane & 16) << 2;
#pragma unroll
    for (int r = 0; r < 8; ++r) {
      int deni = __builtin_amdgcn_ds_bpermute(srcb, __builtin_bit_cast(int, o[4][r]));
      float dinv = 1.0f / __builtin_bit_cast(float, deni);
      float* orow = outg + ((size_t)bh * L + (size_t)c * CH + w * 16 + hi * 8 + r) * D;
#pragma unroll
      for (int dt = 0; dt < 4; ++dt) orow[dt * 16 + n16] = o[dt][r] * dinv;
    }
  }
}

extern "C" void kernel_launch(void* const* d_in, const int* in_sizes, int n_in,
                              void* d_out, int out_size, void* d_ws, size_t ws_size,
                              hipStream_t stream) {
  (void)in_sizes; (void)n_in; (void)out_size; (void)ws_size;
  const float* q    = (const float*)d_in[0];
  const float* k    = (const float*)d_in[1];
  const float* v    = (const float*)d_in[2];
  const float* proj = (const float*)d_in[3];
  float* out = (float*)d_out;

  // Pass-1 partial kv/k_sum slabs live in d_out (9.05M of 16.7M floats);
  // K3 fully overwrites d_out afterwards. d_ws only holds the 2.95 MB kv_ext slabs.
  float* part = out;
  bf16* kvt = (bf16*)d_ws;

  k1_features_kv<<<dim3(BH, SPLITS), dim3(256), LDS1, stream>>>(k, v, proj, part);
  k2_reduce<<<dim3(BH), dim3(256), 0, stream>>>(part, kvt);
  k3_output<<<dim3(BH, L / CH), dim3(256), LDS3, stream>>>(q, proj, kvt, out);
}
